// AttentionMechanism_54752243089428
// MI455X (gfx1250) — matrix operand order repaired and checked by reading, not compiled, not run on hardware
//
#include <hip/hip_runtime.h>

// ---------------- Types for CDNA5 WMMA ----------------
typedef __attribute__((ext_vector_type(16))) __bf16 v16bf;
typedef __attribute__((ext_vector_type(8)))  float  v8f;

#define B_DIM 32
#define T_DIM 2000
#define E_DIM 512
#define A_DIM 512
#define OUT_CH 10
#define KSIZE 101
#define KT_STEPS 17            // 16 k-steps of enc (512) + 1 k-step carrying conv rank-10
#define LDS_PITCH 544          // bf16 elements per staged A row (512 enc + 32 conv/zero)
#define BLK_T 64               // t-rows per workgroup (4 tiles of 16)
#define KT_BYTES 32768         // bytes of packed W per k-step (512n * 2h * 16 bf16)

// bf16 round-to-nearest-even conversion
__device__ __forceinline__ unsigned short f2bf(float f) {
  union { float f; unsigned u; } v; v.f = f;
  unsigned r = v.u + 0x7FFFu + ((v.u >> 16) & 1u);
  return (unsigned short)(r >> 16);
}

// CDNA5 async global->LDS copy (tracked by ASYNCcnt). Generic pointers to LDS
// carry the LDS byte offset in their low 32 bits (flat aperture layout), which
// is exactly the VDST operand of GLOBAL_LOAD_ASYNC_TO_LDS_B128.
__device__ __forceinline__ void async_copy_b128(unsigned lds_addr, const void* gaddr) {
  asm volatile("global_load_async_to_lds_b128 %0, %1, off"
               :: "v"(lds_addr), "v"(gaddr) : "memory");
}
__device__ __forceinline__ void wait_async0() {
  asm volatile("s_wait_asynccnt 0" ::: "memory");
}

// ---------------------------------------------------------------------------
// Prep 1: pack W_enc (512x512, [k][n]) and W_fil (10x512) into the bf16
// B-fragment layout: for (kt, n, h) the 16 bf16 values W[kt*32+h*16+i][n]
// are contiguous -> each lane (n = lane&15, h = lane>>4) does two b128 loads,
// and each k-step's whole panel is one contiguous 32KB block (async-friendly).
// wp index = ((kt*512 + n)*2 + h)*16 + i ; total 17*512*32 elements.
// ---------------------------------------------------------------------------
__global__ __launch_bounds__(256) void pack_w_kernel(
    const float* __restrict__ W_enc, const float* __restrict__ W_fil,
    unsigned short* __restrict__ wp) {
  int idx = blockIdx.x * 256 + threadIdx.x;
  if (idx >= KT_STEPS * 512 * 32) return;
  int i  = idx & 15;
  int h  = (idx >> 4) & 1;
  int n  = (idx >> 5) & 511;
  int kt = idx >> 14;
  int kl = h * 16 + i;
  float v = 0.0f;
  if (kt < 16)          v = W_enc[(kt * 32 + kl) * A_DIM + n];
  else if (kl < OUT_CH) v = W_fil[kl * A_DIM + n];
  wp[idx] = f2bf(v);
}

// ---------------------------------------------------------------------------
// Prep 2: bias[b][a] = b_enc[a] + b_dec[a] + b_fil[a] + dec[b]·W_dec[:,a]
// (b_va dropped: softmax is invariant to a per-batch constant shift)
// ---------------------------------------------------------------------------
__global__ __launch_bounds__(256) void bias_kernel(
    const float* __restrict__ dec, const float* __restrict__ W_dec,
    const float* __restrict__ b_enc, const float* __restrict__ b_dec,
    const float* __restrict__ b_fil, float* __restrict__ bias) {
  int idx = blockIdx.x * 256 + threadIdx.x;   // 32*512
  int b = idx >> 9, a = idx & 511;
  float acc = b_enc[a] + b_dec[a] + b_fil[a];
  const float* d = dec + b * E_DIM;
  for (int e = 0; e < E_DIM; ++e) acc += d[e] * W_dec[e * A_DIM + a];
  bias[idx] = acc;
}

// ---------------------------------------------------------------------------
// Main fused kernel: one workgroup per (b, 64-row t-block). 512 threads = 16
// waves arranged as (tt = wave>>2 in 0..3, aq = wave&3 in 0..3):
//   wave owns t-tile tt (16 rows) x a-quarter aq (128 cols = 8 WMMA tiles).
//  - stage 64 enc rows (f32->bf16) + conv-g block into LDS (A side)
//  - double-buffered async global->LDS copy of the 32KB packed-W panel per kt
//  - 17 k-steps x 8 tiles of v_wmma_f32_16x16x32_bf16 per wave
//  - tanh * v_a, lane xor-reduce over N, LDS reduce over a-quarters -> energy
// ---------------------------------------------------------------------------
__global__ __launch_bounds__(512) void energy_kernel(
    const float* __restrict__ enc, const float* __restrict__ aw,
    const float* __restrict__ conv_w, const float* __restrict__ conv_b,
    const unsigned short* __restrict__ wp, const float* __restrict__ bias,
    const float* __restrict__ v_a, float* __restrict__ energy) {
  extern __shared__ char smem[];
  unsigned short* ldsA = (unsigned short*)smem;                    // 64*544*2 = 69632 B
  unsigned short* ldsB = (unsigned short*)(smem + 69632);          // 2*32768  = 65536 B
  float (*ldsE)[BLK_T] = (float (*)[BLK_T])(smem + 69632 + 65536); // 4*64*4   = 1024 B

  const int tid = threadIdx.x;
  const int b   = blockIdx.y;
  const int t0  = blockIdx.x * BLK_T;
  const unsigned ldsB_base = (unsigned)(size_t)ldsB;

  // ---- kick off async copy of packed-W panel kt=0 into buffer 0 ----
  {
    const char* g0 = (const char*)wp;
    for (int j = 0; j < 4; ++j) {
      int off = j * 8192 + tid * 16;           // 4 iters * 512 thr * 16B = 32KB
      async_copy_b128(ldsB_base + off, g0 + off);
    }
  }

  // ---- stage enc rows as bf16 (float4 -> 4x bf16), rows clamped at T-1 ----
  for (int j = 0; j < 16; ++j) {
    int base = (j * 512 + tid) * 4;            // 0..32767
    int row = base >> 9, e = base & 511;
    int trow = t0 + row; if (trow > T_DIM - 1) trow = T_DIM - 1;
    float4 v = *(const float4*)(enc + ((size_t)b * T_DIM + trow) * E_DIM + e);
    union { unsigned short s[4]; uint2 u; } pk;
    pk.s[0] = f2bf(v.x); pk.s[1] = f2bf(v.y);
    pk.s[2] = f2bf(v.z); pk.s[3] = f2bf(v.w);
    *(uint2*)(ldsA + row * LDS_PITCH + e) = pk.u;
  }
  // ---- zero pad of extra k-block (channels 10..31) ----
  for (int z = tid; z < BLK_T * 22; z += 512) {
    int row = z / 22, kk = 10 + z % 22;
    ldsA[row * LDS_PITCH + 512 + kk] = 0;
  }
  // ---- conv g[t][c] = sum_j aw[b][t+j-50]*conv_w[c][j] + conv_b[c] ----
  for (int p = tid; p < BLK_T * OUT_CH; p += 512) {
    int row = p / OUT_CH, c = p % OUT_CH;
    float acc = conv_b[c];
    int tg = t0 + row;
    for (int j = 0; j < KSIZE; ++j) {
      int tt = tg + j - (KSIZE / 2);
      if (tt >= 0 && tt < T_DIM) acc += aw[b * T_DIM + tt] * conv_w[c * KSIZE + j];
    }
    ldsA[row * LDS_PITCH + 512 + c] = f2bf(acc);
  }

  // ---- WMMA main loop ----
  const int wave = tid >> 5;
  const int lane = tid & 31;
  const int lrow = lane & 15;
  const int h    = lane >> 4;
  const int tt   = wave >> 2;        // t-tile 0..3
  const int aq   = wave & 3;         // a-quarter 0..3
  const int abase = aq * 128;

  typedef union { uint4 u[2]; v16bf v; } frag_t;

  v8f acc[8];
  float vav[8];
  for (int tile = 0; tile < 8; ++tile) {
    int n = abase + tile * 16 + lrow;
    float bv = bias[b * A_DIM + n];
    vav[tile] = v_a[n];
    for (int c = 0; c < 8; ++c) acc[tile][c] = bv;
  }

  const unsigned short* aRow = ldsA + (tt * 16 + lrow) * LDS_PITCH;

  for (int kt = 0; kt < KT_STEPS; ++kt) {
    // make buffer (kt&1) visible to all waves: own asyncs done + barrier.
    // The barrier also guarantees every wave finished its LDS reads of the
    // buffer we are about to overwrite (DScnt is drained before s_barrier).
    wait_async0();
    __syncthreads();
    if (kt + 1 < KT_STEPS) {
      const char* g = (const char*)wp + (size_t)(kt + 1) * KT_BYTES;
      unsigned dstb = ldsB_base + ((kt + 1) & 1) * KT_BYTES;
      for (int j = 0; j < 4; ++j) {
        int off = j * 8192 + tid * 16;
        async_copy_b128(dstb + off, g + off);
      }
    }
    frag_t af;
    const unsigned short* ap = aRow + kt * 32 + h * 16;
    af.u[0] = *(const uint4*)(ap);        // K local 0..7   (16..23 for h=1)
    af.u[1] = *(const uint4*)(ap + 8);    // K local 8..15  (24..31 for h=1)
    const unsigned short* bbuf = ldsB + (kt & 1) * (KT_BYTES / 2);
#pragma unroll
    for (int tile = 0; tile < 8; ++tile) {
      int n = abase + tile * 16 + lrow;
      const uint4* bp = (const uint4*)(bbuf + (n * 2 + h) * 16);
      frag_t bq;
      bq.u[0] = bp[0];
      bq.u[1] = bp[1];
      acc[tile] = __builtin_amdgcn_wmma_f32_16x16x32_bf16(
          false, af.v, false, bq.v, (short)0, acc[tile], false, false);
    }
  }

  // ---- tanh * v_a, reduce over N (16-lane xor shuffles), then a-quarters ----
  float s[8];
  for (int c = 0; c < 8; ++c) s[c] = 0.0f;
  for (int tile = 0; tile < 8; ++tile)
    for (int c = 0; c < 8; ++c)
      s[c] += tanhf(acc[tile][c]) * vav[tile];
  for (int off = 1; off < 16; off <<= 1)
    for (int c = 0; c < 8; ++c)
      s[c] += __shfl_xor(s[c], off, 32);
  if (lane == 0)  for (int c = 0; c < 8; ++c) ldsE[aq][tt * 16 + c]     = s[c];
  if (lane == 16) for (int c = 0; c < 8; ++c) ldsE[aq][tt * 16 + 8 + c] = s[c];
  __syncthreads();
  if (tid < BLK_T) {
    int t = t0 + tid;
    if (t < T_DIM) {
      float e = ldsE[0][tid] + ldsE[1][tid] + ldsE[2][tid] + ldsE[3][tid];
      energy[b * T_DIM + t] = e;
    }
  }
}

// ---------------------------------------------------------------------------
// Softmax over T per batch; writes attn into d_out's attn region.
// ---------------------------------------------------------------------------
__global__ __launch_bounds__(256) void softmax_kernel(
    const float* __restrict__ energy, float* __restrict__ attn) {
  __shared__ float red[256];
  int b = blockIdx.x, tid = threadIdx.x;
  const float* e = energy + b * T_DIM;
  float m = -1e30f;
  for (int t = tid; t < T_DIM; t += 256) m = fmaxf(m, e[t]);
  red[tid] = m; __syncthreads();
  for (int s2 = 128; s2 > 0; s2 >>= 1) {
    if (tid < s2) red[tid] = fmaxf(red[tid], red[tid + s2]);
    __syncthreads();
  }
  float m0 = red[0];
  __syncthreads();
  float sum = 0.0f;
  for (int t = tid; t < T_DIM; t += 256) sum += __expf(e[t] - m0);
  red[tid] = sum; __syncthreads();
  for (int s2 = 128; s2 > 0; s2 >>= 1) {
    if (tid < s2) red[tid] += red[tid + s2];
    __syncthreads();
  }
  float inv = 1.0f / red[0];
  for (int t = tid; t < T_DIM; t += 256)
    attn[b * T_DIM + t] = __expf(e[t] - m0) * inv;
}

// ---------------------------------------------------------------------------
// Context GEMV: partial sums over t-chunks (deterministic, no float atomics).
// grid (8 chunks, 32 batches), 512 threads = one per e. Second enc pass is
// L2-resident (131 MB < 192 MB L2).
// ---------------------------------------------------------------------------
#define TCHUNK 250
__global__ __launch_bounds__(512) void ctx_partial_kernel(
    const float* __restrict__ enc, const float* __restrict__ attn,
    float* __restrict__ pc) {
  int b = blockIdx.y, ch = blockIdx.x, e = threadIdx.x;
  int tstart = ch * TCHUNK;
  const float* encb = enc + ((size_t)b * T_DIM + tstart) * E_DIM + e;
  const float* ab   = attn + b * T_DIM + tstart;
  float a0 = 0.0f, a1 = 0.0f;
  for (int t = 0; t < TCHUNK; t += 2) {
    a0 += ab[t]     * encb[(size_t)t * E_DIM];
    a1 += ab[t + 1] * encb[(size_t)(t + 1) * E_DIM];
  }
  pc[((b * 8 + ch) * E_DIM) + e] = a0 + a1;
}

__global__ __launch_bounds__(256) void ctx_reduce_kernel(
    const float* __restrict__ pc, float* __restrict__ ctx) {
  int idx = blockIdx.x * 256 + threadIdx.x;   // 32*512
  int b = idx >> 9, e = idx & 511;
  float acc = 0.0f;
  for (int c = 0; c < 8; ++c) acc += pc[((b * 8 + c) * E_DIM) + e];
  ctx[idx] = acc;
}

// ---------------------------------------------------------------------------
extern "C" void kernel_launch(void* const* d_in, const int* in_sizes, int n_in,
                              void* d_out, int out_size, void* d_ws, size_t ws_size,
                              hipStream_t stream) {
  const float* enc    = (const float*)d_in[0];   // (32,2000,512)
  const float* dec    = (const float*)d_in[1];   // (32,1,512)
  const float* aw     = (const float*)d_in[2];   // (32,2000)
  const float* W_enc  = (const float*)d_in[3];   // (512,512)
  const float* b_enc  = (const float*)d_in[4];   // (512)
  const float* W_dec  = (const float*)d_in[5];   // (512,512)
  const float* b_dec  = (const float*)d_in[6];   // (512)
  const float* conv_w = (const float*)d_in[7];   // (10,1,101)
  const float* conv_b = (const float*)d_in[8];   // (10)
  const float* W_fil  = (const float*)d_in[9];   // (10,512)
  const float* b_fil  = (const float*)d_in[10];  // (512)
  const float* v_a    = (const float*)d_in[11];  // (512)
  // d_in[12] = b_va : softmax-invariant, unused.

  float* out_ctx  = (float*)d_out;                       // 32*512
  float* out_attn = (float*)d_out + B_DIM * E_DIM;       // 32*2000

  char* ws = (char*)d_ws;
  unsigned short* wp = (unsigned short*)(ws + 0);        // 557056 B
  float* bias   = (float*)(ws + 557056);                 //  65536 B
  float* energy = (float*)(ws + 622592);                 // 256000 B
  float* pc     = (float*)(ws + 878592);                 // 524288 B

  pack_w_kernel<<<(KT_STEPS * 512 * 32 + 255) / 256, 256, 0, stream>>>(W_enc, W_fil, wp);
  bias_kernel<<<(B_DIM * A_DIM) / 256, 256, 0, stream>>>(dec, W_dec, b_enc, b_dec, b_fil, bias);

  const size_t smem_bytes = 69632 + 65536 + 1024;  // A tile + double-buffered B + reduce
  energy_kernel<<<dim3((T_DIM + BLK_T - 1) / BLK_T, B_DIM), 512, smem_bytes, stream>>>(
      enc, aw, conv_w, conv_b, wp, bias, v_a, energy);

  softmax_kernel<<<B_DIM, 256, 0, stream>>>(energy, out_attn);
  ctx_partial_kernel<<<dim3(8, B_DIM), 512, 0, stream>>>(enc, out_attn, pc);
  ctx_reduce_kernel<<<(B_DIM * E_DIM) / 256, 256, 0, stream>>>(pc, out_ctx);
}
